// TritonKroneckerLinear_32847909880415
// MI455X (gfx1250) — matrix operand order repaired and checked by reading, not compiled, not run on hardware
//
#include <hip/hip_runtime.h>
#include <hip/hip_bf16.h>

// Y = X @ kron(A,B)^T + bias, factored form.
// M=16384, A:(1024,1024), B:(4,4), x:(16384,4096) f32, y:(16384,4096) f32.
// Heavy part = 4 batched GEMMs (one per n in 0..3):  Y_n = T_n @ A^T,
// with T_n[m,k] = sum_j x[m,4k+j] * B[n,j] fused into the tile load.
// bf16 WMMA (v_wmma_f32_16x16x32_bf16) with f32 accumulation.
// Tile: 64 m x 128 p x 4 n per workgroup; 8 waves (4 m-waves x 2 p-waves);
// 16 WMMAs per wave per K-step (K-step = 32).

typedef __attribute__((ext_vector_type(16))) __bf16 bf16x16;
typedef __attribute__((ext_vector_type(8)))  __bf16 bf16x8;
typedef __attribute__((ext_vector_type(2)))  __bf16 bf16x2;
typedef __attribute__((ext_vector_type(8)))  float  f32x8;

#define BM 64
#define BP 128
#define BK 32
#define AK 1024   // inner contraction length (a_k)
#define ANN 1024  // a_n
#define XC 4096   // x columns = a_k*b_k
#define YC 4096   // y columns = a_n*b_n

// Pack two f32 into one dword of two bf16 (RNE). Prefer the hw packed cvt.
static __device__ __forceinline__ unsigned int pk_bf16(float a, float b) {
#if __has_builtin(__builtin_amdgcn_cvt_pk_bf16_f32)
    bf16x2 v = __builtin_amdgcn_cvt_pk_bf16_f32(a, b);
    return __builtin_bit_cast(unsigned int, v);
#else
    unsigned int ua = __builtin_bit_cast(unsigned int, a);
    unsigned int ub = __builtin_bit_cast(unsigned int, b);
    ua += 0x7FFFu + ((ua >> 16) & 1u);
    ub += 0x7FFFu + ((ub >> 16) & 1u);
    return (ua >> 16) | (ub & 0xFFFF0000u);   // lowers to v_perm-style merge
#endif
}

__global__ __launch_bounds__(256) void kron_linear_wmma(
    const float* __restrict__ x, const float* __restrict__ A,
    const float* __restrict__ Bm, const float* __restrict__ bias,
    float* __restrict__ y)
{
    // t tile: [n][m][k] bf16, rows of 32 k = 64B (A-frag source)  -> 16 KB
    // At tile: [k][p] bf16, rows of 128 p = 256B (B-frag source)  ->  8 KB
    __shared__ __align__(32) __bf16 lds_t[4][BM][BK];
    __shared__ __align__(32) __bf16 lds_a[BK][BP];

    const int tid  = threadIdx.x;
    const int wid  = tid >> 5;        // 8 waves
    const int lane = tid & 31;
    const int mw   = wid & 3;         // 4 m-waves -> 16 rows each
    const int pw   = wid >> 2;        // 2 p-waves -> 64 cols each
    const int half = lane >> 4;
    const int l15  = lane & 15;

    // p on blockIdx.x so the 8 p-blocks sharing an m-band of x run
    // adjacently -> x band re-reads are served from the 192MB L2.
    const int p0 = blockIdx.x * BP;
    const int m0 = blockIdx.y * BM;

    // 4x4 B factor, row-major (scalarizes into SGPRs)
    float Bv[16];
#pragma unroll
    for (int i = 0; i < 16; ++i) Bv[i] = Bm[i];

    f32x8 acc[4][4];
#pragma unroll
    for (int n = 0; n < 4; ++n)
#pragma unroll
        for (int ps = 0; ps < 4; ++ps) {
            f32x8 z = {0.f,0.f,0.f,0.f,0.f,0.f,0.f,0.f};
            acc[n][ps] = z;
        }

    const int xm2 = tid >> 4;   // 0..15, combined with i*16 -> m 0..63
    const int xkp = tid & 15;   // k-pair index within K-step (k = 2*xkp)
    const int app = tid >> 3;   // A: p-pair index base 0..31
    const int akq = (tid & 7) * 4;  // A: k-quad within K-step

    for (int k0 = 0; k0 < AK; k0 += BK) {
        // ---- stage x tile: fuse 4x4 B contraction, pack pairs to bf16x2 ----
#pragma unroll
        for (int i = 0; i < 4; ++i) {
            const int m = i * 16 + xm2;
            const float* px = x + (size_t)(m0 + m) * XC
                                + (size_t)(k0 + 2 * xkp) * 4;
            const float4 q0 = *(const float4*)(px);
            const float4 q1 = *(const float4*)(px + 4);
#pragma unroll
            for (int n = 0; n < 4; ++n) {
                const float t0 = q0.x*Bv[4*n] + q0.y*Bv[4*n+1]
                               + q0.z*Bv[4*n+2] + q0.w*Bv[4*n+3];
                const float t1 = q1.x*Bv[4*n] + q1.y*Bv[4*n+1]
                               + q1.z*Bv[4*n+2] + q1.w*Bv[4*n+3];
                // one conflict-free ds_store_b32 per (n, k-pair)
                ((unsigned int*)&lds_t[n][m][0])[xkp] = pk_bf16(t0, t1);
            }
        }
        // ---- stage A tile transposed (At[k][p]): pair-pack across p ----
#pragma unroll
        for (int i = 0; i < 2; ++i) {
            const int pp = i * 32 + app;          // p-pair index 0..63
            const int p  = 2 * pp;
            const float* pa = A + (size_t)(p0 + p) * AK + (k0 + akq);
            const float4 q0 = *(const float4*)(pa);        // row p
            const float4 q1 = *(const float4*)(pa + AK);   // row p+1
            ((unsigned int*)&lds_a[akq + 0][0])[pp] = pk_bf16(q0.x, q1.x);
            ((unsigned int*)&lds_a[akq + 1][0])[pp] = pk_bf16(q0.y, q1.y);
            ((unsigned int*)&lds_a[akq + 2][0])[pp] = pk_bf16(q0.z, q1.z);
            ((unsigned int*)&lds_a[akq + 3][0])[pp] = pk_bf16(q0.w, q1.w);
        }
        __syncthreads();

        // prefetch next x K-step into L2 (global_prefetch_b8)
        if (k0 + BK < AK) {
            __builtin_prefetch(x + (size_t)(m0 + xm2) * XC
                                 + (size_t)(k0 + BK + 2 * xkp) * 4, 0, 1);
        }

        // ---- B-frags: lane = p column; 16 K values (table order) ----
        bf16x16 bfrag[4];
#pragma unroll
        for (int ps = 0; ps < 4; ++ps) {
            const int pc = pw * 64 + ps * 16 + l15;
#pragma unroll
            for (int s = 0; s < 16; ++s) {
                const int kk = (s < 8) ? (half * 8 + s) : (16 + half * 8 + (s - 8));
                bfrag[ps][s] = lds_a[kk][pc];
            }
        }

        // ---- A-frags + 16 WMMAs per wave per K-step ----
#pragma unroll
        for (int n = 0; n < 4; ++n) {
            const __bf16* tr = &lds_t[n][mw * 16 + l15][0];
            // lane half h: VGPR0-3 = K h*8..h*8+7, VGPR4-7 = K 16+h*8..+7
            const bf16x8 lo = *(const bf16x8*)(tr + half * 8);
            const bf16x8 hi = *(const bf16x8*)(tr + 16 + half * 8);
            const bf16x16 afrag = __builtin_shufflevector(
                lo, hi, 0,1,2,3,4,5,6,7,8,9,10,11,12,13,14,15);
#pragma unroll
            for (int ps = 0; ps < 4; ++ps) {
                acc[n][ps] = __builtin_amdgcn_wmma_f32_16x16x32_bf16(
                    /*neg_a=*/false, afrag, /*neg_b=*/false, bfrag[ps],
                    /*c_mod=*/(short)0, acc[n][ps],
                    /*reuse_a=*/false, /*reuse_b=*/false);
            }
        }
        __syncthreads();
    }

    // ---- epilogue: pack 4 n-planes per p into float4, add bias, store ----
#pragma unroll
    for (int ps = 0; ps < 4; ++ps) {
        const int pcol = p0 + pw * 64 + ps * 16 + l15;        // global p
        const float4 b4 = *(const float4*)(bias + (size_t)pcol * 4);
#pragma unroll
        for (int r = 0; r < 8; ++r) {
            const int row = m0 + mw * 16 + half * 8 + r;      // C/D layout
            float4 o;
            o.x = acc[0][ps][r] + b4.x;
            o.y = acc[1][ps][r] + b4.y;
            o.z = acc[2][ps][r] + b4.z;
            o.w = acc[3][ps][r] + b4.w;
            *(float4*)(y + (size_t)row * YC + (size_t)pcol * 4) = o;
        }
    }
}

extern "C" void kernel_launch(void* const* d_in, const int* in_sizes, int n_in,
                              void* d_out, int out_size, void* d_ws, size_t ws_size,
                              hipStream_t stream) {
    const float* x    = (const float*)d_in[0];   // (16384, 4096)
    const float* A    = (const float*)d_in[1];   // (1024, 1024)
    const float* B    = (const float*)d_in[2];   // (4, 4)
    const float* bias = (const float*)d_in[3];   // (4096,)
    float* y = (float*)d_out;                    // (16384, 4096)

    dim3 grid(ANN / BP /*8, p-blocks (fast axis -> L2 reuse of x)*/,
              16384 / BM /*256, m-blocks*/);
    kron_linear_wmma<<<grid, 256, 0, stream>>>(x, A, B, bias, y);
}
